// GCN_84370337562979
// MI455X (gfx1250) — compile-verified, matrix operand order checked by reading, not run on hardware
//
#include <hip/hip_runtime.h>
#include <math.h>

// ---------------------------------------------------------------------------
// GCN (3 layers) for gfx1250 / MI455X.
//   N=50000 nodes, E=400000 edges, D=256 features.
//   GEMM via V_WMMA_F32_16X16X32_F16 (f16 in, f32 accumulate), weights
//   pre-packed once per layer into the B-fragment lane layout (f16).
//   Edge aggregation via native fp32 global atomics (working set < 192MB L2).
// ---------------------------------------------------------------------------

typedef __attribute__((ext_vector_type(16))) _Float16 v16h;
typedef __attribute__((ext_vector_type(8)))  float    v8f;

#define DIM 256

__device__ __forceinline__ float atomAddF(float* p, float v) {
  return unsafeAtomicAdd(p, v);   // lowers to global_atomic_add_f32
}

// ---------------------------------------------------------------------------
__global__ void zero_kernel(float* __restrict__ p, int n) {
  int i = blockIdx.x * blockDim.x + threadIdx.x;
  if (i < n) p[i] = 0.0f;
}

// degrees: deg_out[row[e]] += 1 ; deg_in[col[e]] += 1
__global__ void deg_kernel(const int* __restrict__ row, const int* __restrict__ col,
                           float* __restrict__ deg_out, float* __restrict__ deg_in, int E) {
  int e = blockIdx.x * blockDim.x + threadIdx.x;
  if (e < E) {
    atomAddF(&deg_out[row[e]], 1.0f);
    atomAddF(&deg_in[col[e]], 1.0f);
  }
}

// in-place: d[i] = rsqrt(max(d[i], 1))
__global__ void rsqrt_kernel(float* __restrict__ d, int n) {
  int i = blockIdx.x * blockDim.x + threadIdx.x;
  if (i < n) {
    float v = d[i];
    v = v < 1.0f ? 1.0f : v;
    d[i] = rsqrtf(v);
  }
}

// per-edge scatter: agg[col[e], :] += h[row[e], :] * dinv_out[row[e]]
// one thread per (edge, 4-float chunk): E*64 threads.
__global__ __launch_bounds__(256)
void scatter_kernel(const float* __restrict__ h, const float* __restrict__ dinv_out,
                    const int* __restrict__ row, const int* __restrict__ col,
                    float* __restrict__ agg, int E) {
  int idx = blockIdx.x * blockDim.x + threadIdx.x;
  int e  = idx >> 6;
  int c4 = (idx & 63) << 2;
  if (e < E) {
    int r = row[e];
    int c = col[e];
    float s = dinv_out[r];
    const float4 hv = *(const float4*)(h + r * DIM + c4);
    float* dst = agg + c * DIM + c4;
    atomAddF(dst + 0, hv.x * s);
    atomAddF(dst + 1, hv.y * s);
    atomAddF(dst + 2, hv.z * s);
    atomAddF(dst + 3, hv.w * s);
  }
}

// ---------------------------------------------------------------------------
// Pre-pack W (f32, [K=256][N=256] row-major) into f16 B-fragment layout:
// pack element index t = (((tile_n*8 + kc)*32 + lane)*16 + elem),
// elem -> (v = elem>>1, p = elem&1), per ISA 7.12.2 B layout:
//   K = kc*32 + 16*(v>=4) + 8*(lane>=16) + 2*(v&3) + p ;  col = tile_n*16 + (lane&15)
// So each lane's full v16h fragment is 32 contiguous bytes.
__global__ __launch_bounds__(256)
void pack_w_kernel(const float* __restrict__ W, _Float16* __restrict__ pack) {
  int t = blockIdx.x * blockDim.x + threadIdx.x;   // 0 .. 65535
  int elem = t & 15;
  int lane = (t >> 4) & 31;
  int kc   = (t >> 9) & 7;
  int tn   = t >> 12;
  int v = elem >> 1, p = elem & 1;
  int half = lane >> 4, mn = lane & 15;
  int kb = kc * 32 + ((v & 4) << 2) + (half << 3) + ((v & 3) << 1) + p;
  int gn = tn * 16 + mn;
  pack[t] = (_Float16)W[kb * DIM + gn];
}

// ---------------------------------------------------------------------------
// WMMA GEMM + epilogue:
//   out[m,n] = (sum_k A[m,k]*W[k,n] + bias[n]) * dinv_in[m]  (+ resid[m,n])
// One wave per 16x16 output tile; K fully unrolled: 8x v_wmma_f32_16x16x32_f16.
// A fragment per lane = two contiguous 8-float runs -> 4x b128 loads;
// B fragment = one 32-byte packed load.
// ---------------------------------------------------------------------------
template <bool HAS_RESID>
__global__ __launch_bounds__(128)
void gemm_wmma_kernel(const float* __restrict__ A, const _Float16* __restrict__ Wp,
                      const float* __restrict__ bias, const float* __restrict__ dinv_in,
                      const float* __restrict__ resid, float* __restrict__ out) {
  const int wave   = blockIdx.x * (blockDim.x >> 5) + (threadIdx.x >> 5);
  const int lane   = threadIdx.x & 31;
  const int tile_m = wave >> 4;    // 0 .. N/16-1
  const int tile_n = wave & 15;    // 0 .. 15
  const int half   = lane >> 4;
  const int mn     = lane & 15;

  const int gm_a = tile_m * 16 + mn;   // A row for this lane
  const int gn   = tile_n * 16 + mn;   // B/C/D column for this lane
  const float* Arow = A + gm_a * DIM + half * 8;
  const v16h*  Bp   = (const v16h*)Wp + tile_n * 256 + lane;

  v8f c = {};
#pragma unroll
  for (int kc = 0; kc < 8; ++kc) {
    const float* ap = Arow + kc * 32;
    const float4 q0 = *(const float4*)(ap);        // K = k0+8h .. +3
    const float4 q1 = *(const float4*)(ap + 4);    // K = k0+8h+4 .. +7
    const float4 q2 = *(const float4*)(ap + 16);   // K = k0+16+8h .. +3
    const float4 q3 = *(const float4*)(ap + 20);   // K = k0+16+8h+4 .. +7
    v16h a;
    a[0]  = (_Float16)q0.x;  a[1]  = (_Float16)q0.y;
    a[2]  = (_Float16)q0.z;  a[3]  = (_Float16)q0.w;
    a[4]  = (_Float16)q1.x;  a[5]  = (_Float16)q1.y;
    a[6]  = (_Float16)q1.z;  a[7]  = (_Float16)q1.w;
    a[8]  = (_Float16)q2.x;  a[9]  = (_Float16)q2.y;
    a[10] = (_Float16)q2.z;  a[11] = (_Float16)q2.w;
    a[12] = (_Float16)q3.x;  a[13] = (_Float16)q3.y;
    a[14] = (_Float16)q3.z;  a[15] = (_Float16)q3.w;
    const v16h b = Bp[kc * 32];
    c = __builtin_amdgcn_wmma_f32_16x16x32_f16(false, a, false, b,
                                               (short)0, c, false, false);
  }

  const float bn = bias[gn];
#pragma unroll
  for (int r = 0; r < 8; ++r) {
    const int gm = tile_m * 16 + r + half * 8;
    float val = c[r] + bn;
    val *= dinv_in[gm];
    if constexpr (HAS_RESID) val += resid[gm * DIM + gn];
    out[gm * DIM + gn] = val;
  }
}

// ---------------------------------------------------------------------------
// BatchNorm statistics: per-channel sum / sumsq (coalesced, thread = channel)
__global__ __launch_bounds__(256)
void bn_stats_kernel(const float* __restrict__ pre, float* __restrict__ sums,
                     float* __restrict__ sumsq, int n) {
  const int c    = threadIdx.x;       // channel 0..255
  const int row0 = blockIdx.x * 128;
  int rend = row0 + 128;
  if (rend > n) rend = n;
  float s = 0.0f, s2 = 0.0f;
  for (int r = row0; r < rend; ++r) {
    float v = pre[r * DIM + c];
    s  += v;
    s2 += v * v;
  }
  atomAddF(&sums[c], s);
  atomAddF(&sumsq[c], s2);
}

// sums -> mu, sumsq -> 1/sqrt(var+eps)   (biased var, matches BatchNorm1d)
__global__ void bn_finalize_kernel(float* __restrict__ sums, float* __restrict__ sumsq, int n) {
  const int c = threadIdx.x;
  const float inv_n = 1.0f / (float)n;
  float mu  = sums[c] * inv_n;
  float var = sumsq[c] * inv_n - mu * mu;
  sums[c]  = mu;
  sumsq[c] = rsqrtf(var + 1e-5f);
}

// in-place normalize + ReLU
__global__ __launch_bounds__(256)
void bn_apply_kernel(float* __restrict__ buf, const float* __restrict__ mu,
                     const float* __restrict__ inv_std, const float* __restrict__ gamma,
                     const float* __restrict__ beta, int nd) {
  int i = blockIdx.x * blockDim.x + threadIdx.x;
  if (i < nd) {
    const int c = i & (DIM - 1);
    float v = (buf[i] - mu[c]) * inv_std[c] * gamma[c] + beta[c];
    buf[i] = fmaxf(v, 0.0f);
  }
}

// ---------------------------------------------------------------------------
extern "C" void kernel_launch(void* const* d_in, const int* in_sizes, int n_in,
                              void* d_out, int out_size, void* d_ws, size_t ws_size,
                              hipStream_t stream) {
  const float* x  = (const float*)d_in[0];
  const int*   ei = (const int*)d_in[1];
  const float* W0 = (const float*)d_in[2];
  const float* b0 = (const float*)d_in[3];
  const float* g0 = (const float*)d_in[4];
  const float* be0= (const float*)d_in[5];
  const float* W1 = (const float*)d_in[6];
  const float* b1 = (const float*)d_in[7];
  const float* g1 = (const float*)d_in[8];
  const float* be1= (const float*)d_in[9];
  const float* W2 = (const float*)d_in[10];
  const float* b2 = (const float*)d_in[11];
  float* out = (float*)d_out;

  const int N = in_sizes[0] / DIM;   // 50000
  const int E = in_sizes[1] / 2;     // 400000
  const int ND = N * DIM;
  const int* row = ei;
  const int* col = ei + E;

  // workspace layout (floats)
  float* ws       = (float*)d_ws;
  float* dinv_out = ws;                       // N
  float* dinv_in  = ws + N;                   // N
  float* stats    = ws + 2 * N;               // 512 (sums | sumsq)
  float* agg      = ws + 2 * N + 512;         // N*D
  float* bufA     = agg + (size_t)ND;         // N*D
  _Float16* wp0   = (_Float16*)(bufA + (size_t)ND);  // 64K halves each, 32B aligned
  _Float16* wp1   = wp0 + 65536;
  _Float16* wp2   = wp1 + 65536;

  const int zb = 256;
  auto nb = [](int n, int b) { return (n + b - 1) / b; };

  const int scat_blocks = (E * 64) / 256;               // 100000
  const int gemm_blocks = (N / 16) * 16 / 4;            // 12500 (4 waves/block)
  const int bn_blocks   = nb(N, 128);
  const int pack_blocks = 65536 / 256;

  // ---- degrees + weight packing (once) ----
  zero_kernel<<<nb(2 * N, zb), zb, 0, stream>>>(dinv_out, 2 * N);
  deg_kernel<<<nb(E, zb), zb, 0, stream>>>(row, col, dinv_out, dinv_in, E);
  rsqrt_kernel<<<nb(2 * N, zb), zb, 0, stream>>>(dinv_out, 2 * N);
  pack_w_kernel<<<pack_blocks, 256, 0, stream>>>(W0, wp0);
  pack_w_kernel<<<pack_blocks, 256, 0, stream>>>(W1, wp1);
  pack_w_kernel<<<pack_blocks, 256, 0, stream>>>(W2, wp2);

  // ---- layer 0: in = x, temp out = d_out ----
  zero_kernel<<<nb(ND, zb), zb, 0, stream>>>(agg, ND);
  scatter_kernel<<<scat_blocks, 256, 0, stream>>>(x, dinv_out, row, col, agg, E);
  gemm_wmma_kernel<true><<<gemm_blocks, 128, 0, stream>>>(agg, wp0, b0, dinv_in, x, out);
  zero_kernel<<<2, zb, 0, stream>>>(stats, 512);
  bn_stats_kernel<<<bn_blocks, 256, 0, stream>>>(out, stats, stats + 256, N);
  bn_finalize_kernel<<<1, 256, 0, stream>>>(stats, stats + 256, N);
  bn_apply_kernel<<<nb(ND, zb), zb, 0, stream>>>(out, stats, stats + 256, g0, be0, ND);

  // ---- layer 1: in = d_out, out = bufA ----
  zero_kernel<<<nb(ND, zb), zb, 0, stream>>>(agg, ND);
  scatter_kernel<<<scat_blocks, 256, 0, stream>>>(out, dinv_out, row, col, agg, E);
  gemm_wmma_kernel<true><<<gemm_blocks, 128, 0, stream>>>(agg, wp1, b1, dinv_in, out, bufA);
  zero_kernel<<<2, zb, 0, stream>>>(stats, 512);
  bn_stats_kernel<<<bn_blocks, 256, 0, stream>>>(bufA, stats, stats + 256, N);
  bn_finalize_kernel<<<1, 256, 0, stream>>>(stats, stats + 256, N);
  bn_apply_kernel<<<nb(ND, zb), zb, 0, stream>>>(bufA, stats, stats + 256, g1, be1, ND);

  // ---- layer 2: in = bufA, final out = d_out (no resid/BN/ReLU) ----
  zero_kernel<<<nb(ND, zb), zb, 0, stream>>>(agg, ND);
  scatter_kernel<<<scat_blocks, 256, 0, stream>>>(bufA, dinv_out, row, col, agg, E);
  gemm_wmma_kernel<false><<<gemm_blocks, 128, 0, stream>>>(agg, wp2, b2, dinv_in, nullptr, out);
}
